// Block_34256659153605
// MI455X (gfx1250) — matrix operand order repaired and checked by reading, not compiled, not run on hardware
//
#include <hip/hip_runtime.h>
#include <stdint.h>

// ---------------- problem constants ----------------
static constexpr int Bb  = 8;
static constexpr int Nn  = 2048;
static constexpr int Cc  = 512;
static constexpr int Hh  = 4;
static constexpr int HK  = Cc / Hh;   // 128
static constexpr int HV  = Cc / Hh;   // 128
static constexpr int HID = 4 * Cc;    // 2048
static constexpr long long MN  = (long long)Bb * Nn * Cc;   // activation elements
static constexpr int MTOT = Bb * Nn;                        // 16384 rows

typedef __attribute__((ext_vector_type(16))) __bf16 v16bf;
typedef __attribute__((ext_vector_type(8)))  __bf16 v8bf;
typedef __attribute__((ext_vector_type(8)))  float  v8f;
typedef __attribute__((ext_vector_type(4)))  unsigned int u32x4;
typedef __attribute__((ext_vector_type(4)))  int          i32x4;
typedef __attribute__((ext_vector_type(8)))  int          i32x8;

#if defined(__has_builtin)
#if __has_builtin(__builtin_amdgcn_tensor_load_to_lds)
#define HAVE_TDM 1
#endif
#endif

// ---------------- helpers ----------------
__device__ __forceinline__ __bf16 f2bf(float f) {
  unsigned int u = __builtin_bit_cast(unsigned int, f);
  unsigned int r = u + 0x7FFFu + ((u >> 16) & 1u);   // round-to-nearest-even
  unsigned short h = (unsigned short)(r >> 16);
  return __builtin_bit_cast(__bf16, h);
}
__device__ __forceinline__ void store_val(float* p, float v)  { *p = v; }
__device__ __forceinline__ void store_val(__bf16* p, float v) { *p = f2bf(v); }

__device__ __forceinline__ float blockReduceMax(float v, float* sm) {
  #pragma unroll
  for (int o = 16; o > 0; o >>= 1) v = fmaxf(v, __shfl_xor(v, o, 32));
  int lane = threadIdx.x & 31, w = threadIdx.x >> 5, nw = blockDim.x >> 5;
  if (lane == 0) sm[w] = v;
  __syncthreads();
  if (w == 0) {
    float x = (lane < nw) ? sm[lane] : -3.4e38f;
    #pragma unroll
    for (int o = 16; o > 0; o >>= 1) x = fmaxf(x, __shfl_xor(x, o, 32));
    if (lane == 0) sm[0] = x;
  }
  __syncthreads();
  float r = sm[0];
  __syncthreads();
  return r;
}
__device__ __forceinline__ float blockReduceSum(float v, float* sm) {
  #pragma unroll
  for (int o = 16; o > 0; o >>= 1) v += __shfl_xor(v, o, 32);
  int lane = threadIdx.x & 31, w = threadIdx.x >> 5, nw = blockDim.x >> 5;
  if (lane == 0) sm[w] = v;
  __syncthreads();
  if (w == 0) {
    float x = (lane < nw) ? sm[lane] : 0.f;
    #pragma unroll
    for (int o = 16; o > 0; o >>= 1) x += __shfl_xor(x, o, 32);
    if (lane == 0) sm[0] = x;
  }
  __syncthreads();
  float r = sm[0];
  __syncthreads();
  return r;
}

// ---------------- Tensor Data Mover: 2-D tile (128 rows x 32 bf16) -> LDS ----------------
// Row pitch in LDS becomes 40 bf16 (80 B) via TDM pad: 16 DWORDs data + 4 DWORDs pad.
#ifdef HAVE_TDM
__device__ __forceinline__ void tdm_load_tile(const __bf16* gptr, long long row_stride,
                                              unsigned lds_off) {
  unsigned long long ga = (unsigned long long)(uintptr_t)gptr;
  u32x4 g0;
  g0[0] = 1u;                                            // count=1, user descriptor
  g0[1] = lds_off;                                       // lds_addr
  g0[2] = (unsigned)(ga & 0xFFFFFFFFu);                  // global_addr[31:0]
  g0[3] = (unsigned)((ga >> 32) & 0x1FFFFFFu) | (2u << 30);  // addr[56:32] | type=2
  i32x8 g1;
  // data_size=1 (2B), pad_enable, pad_interval=3 (16 DW), pad_amount=3 (4 DW)
  g1[0] = (int)((1u << 16) | (1u << 20) | (3u << 22) | (3u << 25));
  unsigned td0 = 0x7FFFFFFFu, td1 = 0x7FFFFFFFu;         // huge tensor dims (no clip)
  g1[1] = (int)((td0 & 0xFFFFu) << 16);                  // [63:48] = tensor_dim0 lo
  g1[2] = (int)((td0 >> 16) | ((td1 & 0xFFFFu) << 16));  // dim0 hi | dim1 lo
  g1[3] = (int)((td1 >> 16) | (32u << 16));              // dim1 hi | tile_dim0=32
  g1[4] = (int)128u;                                     // tile_dim1=128, tile_dim2=0
  unsigned long long s0 = (unsigned long long)row_stride;
  g1[5] = (int)(s0 & 0xFFFFFFFFu);                       // dim0_stride[31:0]
  g1[6] = (int)((s0 >> 32) & 0xFFFFu);                   // dim0_stride[47:32]
  g1[7] = 0;
  i32x4 z4 = {0, 0, 0, 0};
#if __clang_major__ >= 23
  i32x8 z8 = {0, 0, 0, 0, 0, 0, 0, 0};
  __builtin_amdgcn_tensor_load_to_lds(g0, g1, z4, z4, z8, 0);
#else
  __builtin_amdgcn_tensor_load_to_lds(g0, g1, z4, z4, 0);
#endif
}
#endif

// ---------------- WMMA GEMM ----------------
// C[M,N] (+bias[N]) = A[M,K](bf16,row-major, row stride lda) * B
//   BNMAJOR=true : B given as Bt[n*ldb + k] (N-major, pre-transposed weights)
//   BNMAJOR=false: B given as B[k*ldb + n]  (K-major, activations)
// batch z: base = (z/GH)*s0 + (z%GH)*s1 for each operand.
template <typename OutT, bool RELU, bool BNMAJOR>
__global__ __launch_bounds__(256) void gemm_bf16(
    const __bf16* __restrict__ A, int lda, long long a0, long long a1,
    const __bf16* __restrict__ Bm, int ldb, long long b0, long long b1,
    OutT* __restrict__ Co, int ldc, long long c0, long long c1,
    const float* __restrict__ bias, int K, int GH) {
  __shared__ __align__(16) __bf16 lA[2][128][40];  // 32 K + 8 pad
  __shared__ __align__(16) __bf16 lB[2][128][40];  // [n][k]

  const int z  = blockIdx.z;
  const long long baseA = (long long)(z / GH) * a0 + (long long)(z % GH) * a1;
  const long long baseB = (long long)(z / GH) * b0 + (long long)(z % GH) * b1;
  const long long baseC = (long long)(z / GH) * c0 + (long long)(z % GH) * c1;
  const int m0 = blockIdx.y * 128;
  const int n0 = blockIdx.x * 128;

  const int tid  = threadIdx.x;
  const int lane = tid & 31;
  const int wave = tid >> 5;
  const int wm = wave & 1;        // 2 waves over M (64 rows each)
  const int wn = wave >> 1;       // 4 waves over N (32 cols each)
  const int half = lane >> 4;
  const int r    = lane & 15;

  v8f acc[4][2];
  #pragma unroll
  for (int i = 0; i < 4; ++i)
    #pragma unroll
    for (int j = 0; j < 2; ++j) acc[i][j] = v8f{};

  const int nk = K >> 5;

  // --- tile staging (buffer bb, k-offset k0) ---
  auto stage = [&](int bb, int k0) {
#ifdef HAVE_TDM
    if (tid < 32) {  // wave 0 issues TDM descriptors (EXEC ignored by tensor ops)
      tdm_load_tile(A + baseA + (long long)m0 * lda + k0, lda,
                    (unsigned)(uintptr_t)&lA[bb][0][0]);
      if (BNMAJOR)
        tdm_load_tile(Bm + baseB + (long long)n0 * ldb + k0, ldb,
                      (unsigned)(uintptr_t)&lB[bb][0][0]);
    }
#else
    {
      int row = tid >> 1, kc = (tid & 1) * 16;
      const uint4* src = (const uint4*)(A + baseA + (long long)(m0 + row) * lda + k0 + kc);
      *(uint4*)&lA[bb][row][kc] = *src;
    }
    if (BNMAJOR) {
      int nr = tid >> 1, kc = (tid & 1) * 16;
      const uint4* src = (const uint4*)(Bm + baseB + (long long)(n0 + nr) * ldb + k0 + kc);
      *(uint4*)&lB[bb][nr][kc] = *src;
    }
#endif
    if (!BNMAJOR) {
      int kk = tid >> 3, nc = (tid & 7) * 16;
      union { uint4 u; __bf16 h[16]; } tmp;
      tmp.u = *(const uint4*)(Bm + baseB + (long long)(k0 + kk) * ldb + n0 + nc);
      #pragma unroll
      for (int e = 0; e < 16; ++e) lB[bb][nc + e][kk] = tmp.h[e];
    }
  };

  // prologue: fill buffer 0
  stage(0, 0);
#ifdef HAVE_TDM
  __builtin_amdgcn_s_wait_tensorcnt(0);
#endif
  __syncthreads();

  for (int it = 0; it < nk; ++it) {
    const int cur = it & 1;
    if (it + 1 < nk) stage(cur ^ 1, (it + 1) << 5);

    // ---- fragments from current buffer ----
    v16bf af[4], bf[2];
    #pragma unroll
    for (int i = 0; i < 4; ++i) {
      int row = wm * 64 + i * 16 + r;
      v8bf lo = *(const v8bf*)&lA[cur][row][half * 8];
      v8bf hi = *(const v8bf*)&lA[cur][row][16 + half * 8];
      af[i] = __builtin_shufflevector(lo, hi, 0,1,2,3,4,5,6,7,8,9,10,11,12,13,14,15);
    }
    #pragma unroll
    for (int j = 0; j < 2; ++j) {
      int col = wn * 32 + j * 16 + r;
      v8bf lo = *(const v8bf*)&lB[cur][col][half * 8];
      v8bf hi = *(const v8bf*)&lB[cur][col][16 + half * 8];
      bf[j] = __builtin_shufflevector(lo, hi, 0,1,2,3,4,5,6,7,8,9,10,11,12,13,14,15);
    }
    #pragma unroll
    for (int i = 0; i < 4; ++i)
      #pragma unroll
      for (int j = 0; j < 2; ++j)
        acc[i][j] = __builtin_amdgcn_wmma_f32_16x16x32_bf16(
            false, af[i], false, bf[j], (short)0, acc[i][j], false, false);

#ifdef HAVE_TDM
    __builtin_amdgcn_s_wait_tensorcnt(0);  // next-buffer DMA complete (0 for waves 1..7)
#endif
    __syncthreads();
  }

  // ---- epilogue ----
  #pragma unroll
  for (int i = 0; i < 4; ++i) {
    #pragma unroll
    for (int j = 0; j < 2; ++j) {
      #pragma unroll
      for (int v = 0; v < 8; ++v) {
        int row = m0 + wm * 64 + i * 16 + half * 8 + v;
        int col = n0 + wn * 32 + j * 16 + r;
        float val = acc[i][j][v];
        if (bias) val += bias[col];
        if (RELU) val = fmaxf(val, 0.f);
        store_val(&Co[baseC + (long long)row * ldc + col], val);
      }
    }
  }
}

// ---------------- elementwise / reduction kernels ----------------
__global__ void k_transpose_bf(const float* __restrict__ W, __bf16* __restrict__ Wt,
                               int K, int Nw) {
  long long idx = (long long)blockIdx.x * blockDim.x + threadIdx.x;
  if (idx >= (long long)K * Nw) return;
  int n = (int)(idx / K), k = (int)(idx - (long long)n * K);
  Wt[idx] = f2bf(W[(long long)k * Nw + n]);
}

__global__ void k_addpos_bf(const float* __restrict__ a, const float* __restrict__ pos,
                            __bf16* __restrict__ o, long long total, int nc) {
  long long i = (long long)blockIdx.x * blockDim.x + threadIdx.x;
  if (i < total) o[i] = f2bf(a[i] + pos[(int)(i % nc)]);
}

__global__ void k_conv_bf(const float* __restrict__ a, __bf16* __restrict__ o,
                          long long total) {
  long long i = (long long)blockIdx.x * blockDim.x + threadIdx.x;
  if (i < total) o[i] = f2bf(a[i]);
}

// softmax over tokens (axis=1) of kp (B,N,C); write transposed (B,H,HK,N) bf16
__global__ __launch_bounds__(256) void k_softmax_tokens(const float* __restrict__ kp,
                                                        __bf16* __restrict__ kT) {
  __shared__ float sm[8];
  int b = blockIdx.x / Cc, c = blockIdx.x % Cc;
  int h = c / HK, kc = c % HK;
  int tid = threadIdx.x;
  float v[Nn / 256];
  float m = -3.4e38f;
  #pragma unroll
  for (int j = 0; j < Nn / 256; ++j) {
    int n = tid + j * 256;
    v[j] = kp[((long long)b * Nn + n) * Cc + c];
    m = fmaxf(m, v[j]);
  }
  m = blockReduceMax(m, sm);
  float s = 0.f;
  #pragma unroll
  for (int j = 0; j < Nn / 256; ++j) { v[j] = __expf(v[j] - m); s += v[j]; }
  s = blockReduceSum(s, sm);
  float inv = 1.f / s;
  long long ob = (((long long)b * Hh + h) * HK + kc) * Nn;
  #pragma unroll
  for (int j = 0; j < Nn / 256; ++j) kT[ob + tid + j * 256] = f2bf(v[j] * inv);
}

// softmax over the HK channels within each head; qp (B,N,C) -> qh bf16 same layout
__global__ __launch_bounds__(128) void k_softmax_ch(const float* __restrict__ qp,
                                                    __bf16* __restrict__ qh) {
  __shared__ float sm[4];
  long long base = (long long)blockIdx.x * Cc;
  int tid = threadIdx.x;
  for (int h = 0; h < Hh; ++h) {
    float x = qp[base + h * HK + tid];
    float m = blockReduceMax(x, sm);
    float e = __expf(x - m);
    float s = blockReduceSum(e, sm);
    qh[base + h * HK + tid] = f2bf(e / s);
  }
}

// out = LayerNorm(a + bres); writes both fp32 and bf16 copies
__global__ __launch_bounds__(256) void k_resid_ln(const float* __restrict__ a,
                                                  const float* __restrict__ bres,
                                                  const float* __restrict__ g,
                                                  const float* __restrict__ beta,
                                                  float* __restrict__ of,
                                                  __bf16* __restrict__ ob) {
  __shared__ float sm[8];
  long long base = (long long)blockIdx.x * Cc;
  int tid = threadIdx.x;
  float t0 = a[base + tid] + bres[base + tid];
  float t1 = a[base + tid + 256] + bres[base + tid + 256];
  float s  = blockReduceSum(t0 + t1, sm);
  float sq = blockReduceSum(t0 * t0 + t1 * t1, sm);
  float mean = s * (1.f / Cc);
  float var  = sq * (1.f / Cc) - mean * mean;
  float inv  = rsqrtf(var + 1e-5f);
  float o0 = (t0 - mean) * inv * g[tid] + beta[tid];
  float o1 = (t1 - mean) * inv * g[tid + 256] + beta[tid + 256];
  of[base + tid] = o0;        ob[base + tid] = f2bf(o0);
  of[base + tid + 256] = o1;  ob[base + tid + 256] = f2bf(o1);
}

// ---------------- host-side orchestration ----------------
template <typename OutT, bool RELU, bool BN>
static void gemm(hipStream_t s,
                 const __bf16* A, int lda, long long a0, long long a1,
                 const __bf16* B, int ldb, long long b0, long long b1,
                 OutT* C, int ldc, long long c0, long long c1,
                 const float* bias, int M, int N, int K, int batches, int GH) {
  dim3 g(N / 128, M / 128, batches);
  gemm_bf16<OutT, RELU, BN><<<g, 256, 0, s>>>(A, lda, a0, a1, B, ldb, b0, b1,
                                              C, ldc, c0, c1, bias, K, GH);
}

struct EScratch {
  float* qp; float* kp;
  __bf16* vp_bf; __bf16* kT; __bf16* qh; __bf16* ctx; __bf16* att_bf;
};

static void run_eattn(hipStream_t s,
                      const __bf16* qin, const __bf16* kin, const __bf16* vin,
                      const __bf16* wq, const float* bq,
                      const __bf16* wk, const float* bk,
                      const __bf16* wv, const float* bv,
                      const __bf16* wr, const float* br,
                      const EScratch& t, float* out) {
  // projections: (16384x512) @ (512x512)
  gemm<float, false, true>(s, qin, Cc, 0, 0, wq, Cc, 0, 0, t.qp, Cc, 0, 0, bq,
                           MTOT, Cc, Cc, 1, 1);
  gemm<float, false, true>(s, kin, Cc, 0, 0, wk, Cc, 0, 0, t.kp, Cc, 0, 0, bk,
                           MTOT, Cc, Cc, 1, 1);
  gemm<__bf16, false, true>(s, vin, Cc, 0, 0, wv, Cc, 0, 0, t.vp_bf, Cc, 0, 0, bv,
                            MTOT, Cc, Cc, 1, 1);
  k_softmax_tokens<<<Bb * Cc, 256, 0, s>>>(t.kp, t.kT);
  k_softmax_ch<<<Bb * Nn, 128, 0, s>>>(t.qp, t.qh);
  // ctx[b,h] = kT(128xN) @ v(Nx128): batched M=128,N=128,K=Nn
  gemm<__bf16, false, false>(s,
      t.kT, Nn, (long long)Hh * HK * Nn, (long long)HK * Nn,
      t.vp_bf, Cc, (long long)Nn * Cc, HV,
      t.ctx, 128, (long long)Hh * 128 * 128, 128 * 128,
      nullptr, 128, 128, Nn, Bb * Hh, Hh);
  // att[b,h] = qh(Nx128) @ ctx(128x128): batched M=Nn,N=128,K=128 -> (B,N,C) bf16
  gemm<__bf16, false, false>(s,
      t.qh, Cc, (long long)Nn * Cc, HK,
      t.ctx, 128, (long long)Hh * 128 * 128, 128 * 128,
      t.att_bf, Cc, (long long)Nn * Cc, HV,
      nullptr, Nn, 128, 128, Bb * Hh, Hh);
  // output projection
  gemm<float, false, true>(s, t.att_bf, Cc, 0, 0, wr, Cc, 0, 0, out, Cc, 0, 0, br,
                           MTOT, Cc, Cc, 1, 1);
}

extern "C" void kernel_launch(void* const* d_in, const int* in_sizes, int n_in,
                              void* d_out, int out_size, void* d_ws, size_t ws_size,
                              hipStream_t stream) {
  (void)in_sizes; (void)n_in; (void)out_size; (void)ws_size;
  const float* x     = (const float*)d_in[0];
  const float* y     = (const float*)d_in[1];
  const float* q     = (const float*)d_in[2];
  const float* pos_x = (const float*)d_in[3];
  const float* pos_y = (const float*)d_in[4];
  const float* Wf[8]  = { (const float*)d_in[5],  (const float*)d_in[7],
                          (const float*)d_in[9],  (const float*)d_in[11],
                          (const float*)d_in[13], (const float*)d_in[15],
                          (const float*)d_in[17], (const float*)d_in[19] };
  const float* Bv[8]  = { (const float*)d_in[6],  (const float*)d_in[8],
                          (const float*)d_in[10], (const float*)d_in[12],
                          (const float*)d_in[14], (const float*)d_in[16],
                          (const float*)d_in[18], (const float*)d_in[20] };
  const float* w1 = (const float*)d_in[21];
  const float* b1 = (const float*)d_in[22];
  const float* w2 = (const float*)d_in[23];
  const float* b2 = (const float*)d_in[24];
  const float* ln_g = (const float*)d_in[25];
  const float* ln_b = (const float*)d_in[26];

  char* wsp = (char*)d_ws;
  auto alloc = [&](size_t bytes) -> void* {
    void* p = (void*)wsp;
    wsp += (bytes + 255) & ~(size_t)255;
    return p;
  };

  // transposed bf16 weights
  __bf16* Wt[8];
  for (int i = 0; i < 8; ++i) Wt[i] = (__bf16*)alloc((size_t)Cc * Cc * 2);
  __bf16* W1t = (__bf16*)alloc((size_t)HID * Cc * 2);   // (2048 x 512)
  __bf16* W2t = (__bf16*)alloc((size_t)Cc * HID * 2);   // (512 x 2048)

  // bf16 activation buffers
  __bf16* bf0 = (__bf16*)alloc(MN * 2);   // q_in
  __bf16* bf1 = (__bf16*)alloc(MN * 2);   // k_in
  __bf16* bf2 = (__bf16*)alloc(MN * 2);   // v_in
  __bf16* bf3 = (__bf16*)alloc(MN * 2);   // vp
  __bf16* bf4 = (__bf16*)alloc(MN * 2);   // kT
  __bf16* bf5 = (__bf16*)alloc(MN * 2);   // qh
  __bf16* bf6 = (__bf16*)alloc(MN * 2);   // att
  __bf16* bf7 = (__bf16*)alloc(MN * 2);   // resid_ln bf16 out

  // fp32 buffers (qp+kp region is reused as the 64MB bf16 MLP hidden)
  float* f_qp  = (float*)alloc(MN * 4);
  float* f_kp  = (float*)alloc(MN * 4);
  float* f_pr  = (float*)alloc(MN * 4);   // attn/mlp projection out
  float* f_sa  = (float*)alloc(MN * 4);   // Osa / ysa
  float* f_ca  = (float*)alloc(MN * 4);   // Oca / yca
  float* f_oo  = (float*)alloc(MN * 4);   // Oo
  __bf16* ctx  = (__bf16*)alloc((size_t)Bb * Hh * 128 * 128 * 2);
  __bf16* hid  = (__bf16*)f_qp;           // 64 MB: overlaps qp+kp (dead there)

  EScratch es{ f_qp, f_kp, bf3, bf4, bf5, ctx, bf6 };

  // weight prep
  for (int i = 0; i < 8; ++i)
    k_transpose_bf<<<(Cc * Cc + 255) / 256, 256, 0, stream>>>(Wf[i], Wt[i], Cc, Cc);
  k_transpose_bf<<<(Cc * HID + 255) / 256, 256, 0, stream>>>(w1, W1t, Cc, HID);
  k_transpose_bf<<<(Cc * HID + 255) / 256, 256, 0, stream>>>(w2, W2t, HID, Cc);

  const int NC = Nn * Cc;
  const int EG = (int)((MN + 255) / 256);

  auto mlp_ln = [&](const __bf16* in_bf, const float* resid, float* out_f, __bf16* out_bf) {
    gemm<__bf16, true, true>(stream, in_bf, Cc, 0, 0, W1t, Cc, 0, 0,
                             hid, HID, 0, 0, b1, MTOT, HID, Cc, 1, 1);
    gemm<float, false, true>(stream, hid, HID, 0, 0, W2t, HID, 0, 0,
                             f_pr, Cc, 0, 0, b2, MTOT, Cc, HID, 1, 1);
    k_resid_ln<<<MTOT, 256, 0, stream>>>(f_pr, resid, ln_g, ln_b, out_f, out_bf);
  };

  // ===== x branch =====
  k_addpos_bf<<<EG, 256, 0, stream>>>(x, pos_x, bf0, MN, NC);   // x+pos (q_in=k_in)
  k_conv_bf<<<EG, 256, 0, stream>>>(x, bf2, MN);                // v_in = x
  run_eattn(stream, bf0, bf0, bf2, Wt[0], Bv[0], Wt[1], Bv[1], Wt[2], Bv[2],
            Wt[3], Bv[3], es, f_pr);
  k_resid_ln<<<MTOT, 256, 0, stream>>>(f_pr, x, ln_g, ln_b, f_sa, bf7);   // Osa

  k_addpos_bf<<<EG, 256, 0, stream>>>(q, pos_x, bf0, MN, NC);        // q+pos_x
  k_addpos_bf<<<EG, 256, 0, stream>>>(f_sa, pos_x, bf1, MN, NC);     // Osa+pos_x
  k_conv_bf<<<EG, 256, 0, stream>>>(f_sa, bf2, MN);                  // v_in = Osa
  run_eattn(stream, bf0, bf1, bf2, Wt[4], Bv[4], Wt[5], Bv[5], Wt[6], Bv[6],
            Wt[7], Bv[7], es, f_pr);
  k_resid_ln<<<MTOT, 256, 0, stream>>>(f_pr, f_sa, ln_g, ln_b, f_ca, bf7); // Oca

  mlp_ln(bf7, f_ca, f_oo, bf7);                                      // Oo

  // ===== y branch =====
  k_addpos_bf<<<EG, 256, 0, stream>>>(y, pos_y, bf0, MN, NC);
  k_conv_bf<<<EG, 256, 0, stream>>>(y, bf2, MN);
  run_eattn(stream, bf0, bf0, bf2, Wt[0], Bv[0], Wt[1], Bv[1], Wt[2], Bv[2],
            Wt[3], Bv[3], es, f_pr);
  k_resid_ln<<<MTOT, 256, 0, stream>>>(f_pr, y, ln_g, ln_b, f_sa, bf7);   // ysa

  k_addpos_bf<<<EG, 256, 0, stream>>>(f_oo, pos_y, bf0, MN, NC);     // Oo+pos_y
  k_addpos_bf<<<EG, 256, 0, stream>>>(f_sa, pos_y, bf1, MN, NC);     // ysa+pos_y
  k_conv_bf<<<EG, 256, 0, stream>>>(f_sa, bf2, MN);                  // v_in = ysa
  run_eattn(stream, bf0, bf1, bf2, Wt[4], Bv[4], Wt[5], Bv[5], Wt[6], Bv[6],
            Wt[7], Bv[7], es, f_pr);
  k_resid_ln<<<MTOT, 256, 0, stream>>>(f_pr, f_sa, ln_g, ln_b, f_ca, bf7); // yca

  // final MLP + LN -> d_out (fp32)
  gemm<__bf16, true, true>(stream, bf7, Cc, 0, 0, W1t, Cc, 0, 0,
                           hid, HID, 0, 0, b1, MTOT, HID, Cc, 1, 1);
  gemm<float, false, true>(stream, hid, HID, 0, 0, W2t, HID, 0, 0,
                           f_pr, Cc, 0, 0, b2, MTOT, Cc, HID, 1, 1);
  k_resid_ln<<<MTOT, 256, 0, stream>>>(f_pr, f_ca, ln_g, ln_b, (float*)d_out, bf7);
}
